// CAMBlock_11673721111098
// MI455X (gfx1250) — matrix operand order repaired and checked by reading, not compiled, not run on hardware
//
#include <hip/hip_runtime.h>

// ---------------- problem constants ----------------
#define NH    8
#define CDIM  256
#define HD    32
#define MLPH  1024
#define TTOK  100352           // 32 * 56 * 56 tokens
#define NWINS 2048             // 32 * 8 * 8 windows
#define SCALE 0.17677669529663688f  // 32^-0.5

typedef _Float16 h16v __attribute__((ext_vector_type(16)));
typedef _Float16 h8v  __attribute__((ext_vector_type(8)));
typedef float    f8v  __attribute__((ext_vector_type(8)));
typedef float    f4v  __attribute__((ext_vector_type(4)));
typedef unsigned int u32x4 __attribute__((ext_vector_type(4)));
typedef int      i32x4 __attribute__((ext_vector_type(4)));
typedef int      i32x8 __attribute__((ext_vector_type(8)));

// LDS (address_space(3)) qualified types so laundered pointers keep DS lowering.
typedef __attribute__((address_space(3))) _Float16 lf16;
typedef __attribute__((address_space(3))) h8v      lh8v;

#if __has_builtin(__builtin_amdgcn_tensor_load_to_lds) && __has_builtin(__builtin_amdgcn_s_wait_tensorcnt)
#define USE_TDM 1
#else
#define USE_TDM 0
#endif

__device__ __forceinline__ h16v cat8(h8v lo, h8v hi) {
  return __builtin_shufflevector(lo, hi, 0,1,2,3,4,5,6,7,8,9,10,11,12,13,14,15);
}
__device__ __forceinline__ f8v wmma_f16(h16v a, h16v b, f8v c) {
  return __builtin_amdgcn_wmma_f32_16x16x32_f16(false, a, false, b, (short)0, c, false, false);
}
// A-operand (16x32 f16) from row-major matrix; `A` points at tile [m=0][k=0].
template <typename PT>
__device__ __forceinline__ h16v load_a(const PT* A, int ld, int lane) {
  int m = lane & 15, hh = lane >> 4;
  const PT* r = A + (size_t)m * ld + hh * 16;
  return cat8(*(const h8v*)(r), *(const h8v*)(r + 8));
}
// B-operand (32x16 f16) where B[k][n] = Wt[n][k]; `Wt` points at tile [n=0][k=0].
template <typename PT>
__device__ __forceinline__ h16v load_bT(const PT* Wt, int ld, int lane) {
  int n = lane & 15, g = lane >> 4;
  const PT* r = Wt + (size_t)n * ld + g * 16;
  return cat8(*(const h8v*)(r), *(const h8v*)(r + 8));
}
// B-operand from an LDS panel in [n][32k] layout, via AS(3) pointer -> ds_load.
__device__ __forceinline__ h16v load_b_lds(const lf16* base, int lane) {
  int n = lane & 15, g = lane >> 4;
  const lf16* r = base + n * 32 + g * 16;
  h8v lo = *(const lh8v*)r;
  h8v hi = *(const lh8v*)(r + 8);
  return cat8(lo, hi);
}

#if USE_TDM
// TDM: async-copy a 64(row) x 32(k) f16 tile, row stride K elements, into LDS.
// D# per CDNA5 ISA (group0: count/lds/global/type,
// group1: data_size=2B, tile 32x64, dim0 stride=K, huge tensor dims -> no OOB).
__device__ __forceinline__ void tdm_load_panel(const _Float16* gsrc, unsigned lds_off, int K) {
  unsigned long long ga = (unsigned long long)(size_t)gsrc;
  u32x4 g0;
  g0[0] = 1u;                                            // count=1, user descriptor
  g0[1] = lds_off;                                       // lds_addr (bytes)
  g0[2] = (unsigned)(ga & 0xFFFFFFFFu);                  // global_addr[31:0]
  g0[3] = (unsigned)((ga >> 32) & 0x01FFFFFFu) | (2u << 30);  // addr[56:32] | type=2
  i32x8 g1;
  g1[0] = (1 << 16);               // workgroup_mask=0, data_size=1 (2 bytes)
  g1[1] = 0;                       // atomic_barrier_addr=0, tensor_dim0[15:0]=0
  g1[2] = 0x4000;                  // tensor_dim0 = 1<<30 (hi16), tensor_dim1[15:0]=0
  g1[3] = 0x4000 | (32 << 16);     // tensor_dim1 = 1<<30 (hi16), tile_dim0 = 32
  g1[4] = 64;                      // tile_dim1 = 64, tile_dim2 = 0
  g1[5] = K;                       // tensor_dim0_stride[31:0] (elements)
  g1[6] = 0;                       // stride hi, tensor_dim1_stride lo
  g1[7] = 0;
  i32x4 z4 = {};
#if defined(__clang_major__) && (__clang_major__ >= 23)
  i32x8 z8 = {};
  __builtin_amdgcn_tensor_load_to_lds(g0, g1, z4, z4, z8, 0);
#else
  __builtin_amdgcn_tensor_load_to_lds(g0, g1, z4, z4, 0);
#endif
}
#endif

// ---------------- weight convert + transpose to f16 [out][in] ----------------
__global__ void k_prep_weights(const float* __restrict__ qw,  const float* __restrict__ kvw,
                               const float* __restrict__ pw,  const float* __restrict__ f1w,
                               const float* __restrict__ f2w,
                               _Float16* __restrict__ qwT, _Float16* __restrict__ kvwT,
                               _Float16* __restrict__ pwT, _Float16* __restrict__ f1T,
                               _Float16* __restrict__ f2T) {
  int idx = blockIdx.x * blockDim.x + threadIdx.x;
  if (idx < 256 * 256)  { int n = idx & 255,  k = idx >> 8;  qwT[n * 256 + k]  = (_Float16)qw[k * 256 + n]; }
  if (idx < 256 * 512)  { int n = idx & 511,  k = idx >> 9;  kvwT[n * 256 + k] = (_Float16)kvw[k * 512 + n]; }
  if (idx < 256 * 256)  { int n = idx & 255,  k = idx >> 8;  pwT[n * 256 + k]  = (_Float16)pw[k * 256 + n]; }
  if (idx < 256 * 1024) { int n = idx & 1023, k = idx >> 10; f1T[n * 256 + k]  = (_Float16)f1w[k * 1024 + n]; }
  if (idx < 1024 * 256) { int n = idx & 255,  k = idx >> 8;  f2T[n * 1024 + k] = (_Float16)f2w[k * 256 + n]; }
}

// ---------------- dynamic position bias MLP + gather ----------------
__device__ void seq16(const float* x, const float* g, const float* b,
                      const float* W, const float* bb, float* y, int nout) {
  float m = 0.f, v = 0.f;
  for (int i = 0; i < 16; ++i) m += x[i];
  m *= (1.f / 16.f);
  for (int i = 0; i < 16; ++i) { float d = x[i] - m; v += d * d; }
  v *= (1.f / 16.f);
  float rs = rsqrtf(v + 1e-5f);
  float t[16];
  for (int i = 0; i < 16; ++i) { float u = (x[i] - m) * rs * g[i] + b[i]; t[i] = u > 0.f ? u : 0.f; }
  for (int o = 0; o < nout; ++o) {
    float s = bb[o];
    for (int i = 0; i < 16; ++i) s += t[i] * W[i * nout + o];
    y[o] = s;
  }
}

__global__ void k_posbias(const float* __restrict__ ppw, const float* __restrict__ ppb,
                          const float* __restrict__ g1, const float* __restrict__ b1,
                          const float* __restrict__ w1, const float* __restrict__ bb1,
                          const float* __restrict__ g2, const float* __restrict__ b2,
                          const float* __restrict__ w2, const float* __restrict__ bb2,
                          const float* __restrict__ g3, const float* __restrict__ b3,
                          const float* __restrict__ w3, const float* __restrict__ bb3,
                          float* __restrict__ rpb) {
  __shared__ float pos[169][NH];
  __shared__ float W1[256], W2[256], W3[128];
  int tid = threadIdx.x;
  if (tid < 256) { W1[tid] = w1[tid]; W2[tid] = w2[tid]; }
  if (tid < 128) W3[tid] = w3[tid];
  __syncthreads();
  if (tid < 169) {
    int i = tid / 13, j = tid % 13;
    float c0 = (float)(i - 6), c1 = (float)(j - 6);
    float x0[16], x1[16], x2[16], x3[NH];
    for (int o = 0; o < 16; ++o) x0[o] = c0 * ppw[o] + c1 * ppw[16 + o] + ppb[o];
    seq16(x0, g1, b1, W1, bb1, x1, 16);
    seq16(x1, g2, b2, W2, bb2, x2, 16);
    seq16(x2, g3, b3, W3, bb3, x3, NH);
    for (int h = 0; h < NH; ++h) pos[tid][h] = x3[h];
  }
  __syncthreads();
  for (int e = tid; e < NH * 64 * 64; e += blockDim.x) {
    int n = e & 63, m2 = (e >> 6) & 63, h = e >> 12;
    float val = 0.f;
    if (m2 < 49 && n < 49) {
      int rm = m2 / 7, cm = m2 % 7, rn = n / 7, cn = n % 7;
      int idx = (rm - rn + 6) * 13 + (cm - cn + 6);
      val = pos[idx][h];
    }
    rpb[e] = val;
  }
}

// ---------------- LayerNorm (wave-per-token) -> f16 ----------------
__global__ void __launch_bounds__(256) k_layernorm(const float* __restrict__ x,
                                                   const float* __restrict__ g,
                                                   const float* __restrict__ b,
                                                   _Float16* __restrict__ out) {
  int w = threadIdx.x >> 5, lane = threadIdx.x & 31;
  int tok = blockIdx.x * 8 + w;
  const float* row = x + (size_t)tok * CDIM + lane * 8;
  f4v v0 = *(const f4v*)row;
  f4v v1 = *(const f4v*)(row + 4);
  float s = 0.f, sq = 0.f;
  for (int i = 0; i < 4; ++i) { s += v0[i] + v1[i]; sq += v0[i]*v0[i] + v1[i]*v1[i]; }
  for (int m = 16; m; m >>= 1) { s += __shfl_xor(s, m, 32); sq += __shfl_xor(sq, m, 32); }
  float mean = s * (1.f / CDIM);
  float var  = sq * (1.f / CDIM) - mean * mean;
  float rs   = rsqrtf(var + 1e-5f);
  h8v o;
  for (int i = 0; i < 4; ++i) {
    o[i]     = (_Float16)((v0[i] - mean) * rs * g[lane * 8 + i]     + b[lane * 8 + i]);
    o[4 + i] = (_Float16)((v1[i] - mean) * rs * g[lane * 8 + 4 + i] + b[lane * 8 + 4 + i]);
  }
  *(h8v*)(out + (size_t)tok * CDIM + lane * 8) = o;
}

// ---------------- WMMA GEMM: 128x64 block tile, 8 waves ----------------
// B panel (64 cols x 32 k, shared by all waves) staged in LDS, double-buffered,
// filled by the Tensor Data Mover (TENSORcnt) when available.
// mode 0: outH = (A@W + bias) * scale  (f16)
// mode 1: outH = gelu_exact(A@W + bias) (f16)
// mode 2: outF = resid + A@W + bias     (f32)
__global__ void __launch_bounds__(256) k_gemm(const _Float16* __restrict__ A,
                                              const _Float16* __restrict__ Wt,
                                              const float* __restrict__ bias,
                                              int N, int K, float scale, int mode,
                                              _Float16* __restrict__ outH,
                                              const float* __restrict__ resid,
                                              float* __restrict__ outF) {
  __shared__ __align__(16) _Float16 Bs[2][64 * 32];  // 2 x 4KB panels, [n][k]
  int tid = threadIdx.x;
  int w = tid >> 5, lane = tid & 31;
  int row0 = blockIdx.x * 128 + w * 16;
  int col0 = blockIdx.y * 64;
  const _Float16* Wpan = Wt + (size_t)col0 * K;

#if USE_TDM
  // The TDM writes LDS through a descriptor the compiler cannot see. Launder
  // the panel pointer through a side-effecting asm so its address escapes
  // (otherwise LLVM proves Bs is never stored and folds the ds_loads to
  // undef). Then cast the laundered generic pointer BACK into address space 3
  // so the panel reads lower as ds_load_b128 (not flat_load_b128).
  _Float16* bsg = &Bs[0][0];
  asm volatile("" : "+v"(bsg) :: "memory");
  const lf16* bs = (const lf16*)bsg;
#define STAGE_B(kc, buf) do { if (w == 0) tdm_load_panel(Wpan + (kc), (unsigned)((buf) * 4096), K); } while (0)
#define SYNC_B()         do { if (w == 0) __builtin_amdgcn_s_wait_tensorcnt((unsigned short)0); \
                              __syncthreads(); asm volatile("" ::: "memory"); } while (0)
#else
  _Float16* bs = &Bs[0][0];
  int bn = tid >> 2, bk = (tid & 3) * 8;  // 256 threads cover 64 rows x 4 chunks
#define STAGE_B(kc, buf) do { *(h8v*)&bs[(buf) * 2048 + bn * 32 + bk] = *(const h8v*)(Wpan + (size_t)bn * K + (kc) + bk); } while (0)
#define SYNC_B()         __syncthreads()
#endif

  int nk = K >> 5;
  STAGE_B(0, 0);
  h16v a_cur = load_a(A + (size_t)row0 * K, K, lane);
  SYNC_B();

  f8v acc[4] = {};
  for (int s = 0; s < nk; ++s) {
    int buf = s & 1;
    if (s + 1 < nk) {
      STAGE_B((s + 1) << 5, buf ^ 1);
      __builtin_prefetch(A + (size_t)row0 * K + ((s + 1) << 5), 0, 1);
    }
    h16v a_nxt = a_cur;
    if (s + 1 < nk) a_nxt = load_a(A + (size_t)row0 * K + ((s + 1) << 5), K, lane);
#pragma unroll
    for (int nt = 0; nt < 4; ++nt) {
#if USE_TDM
      h16v bmat = load_b_lds(bs + buf * 2048 + nt * 512, lane);
#else
      h16v bmat = load_bT(bs + buf * 2048 + nt * 512, 32, lane);
#endif
      acc[nt] = wmma_f16(a_cur, bmat, acc[nt]);
    }
    a_cur = a_nxt;
    SYNC_B();  // panel s+1 ready; all waves done reading buf before it is reused
  }
#undef STAGE_B
#undef SYNC_B

  int n = lane & 15, hl = lane >> 4;
#pragma unroll
  for (int nt = 0; nt < 4; ++nt) {
    int col = col0 + nt * 16 + n;
    float bv = bias[col];
#pragma unroll
    for (int i = 0; i < 8; ++i) {
      size_t row = (size_t)(row0 + hl * 8 + i);
      float v = acc[nt][i] + bv;
      if (mode == 0) {
        outH[row * N + col] = (_Float16)(v * scale);
      } else if (mode == 1) {
        float ge = 0.5f * v * (1.0f + erff(v * 0.70710678118654752f));
        outH[row * N + col] = (_Float16)ge;
      } else {
        outF[row * N + col] = resid[row * N + col] + v;
      }
    }
  }
}

// ---------------- fused window attention + proj + residual ----------------
// grid: 2048 windows; block: 256 threads = 8 waves; wave w == head h.
__global__ void __launch_bounds__(256) k_attn(const _Float16* __restrict__ Qh,
                                              const _Float16* __restrict__ KV,
                                              const float* __restrict__ rpb,
                                              const _Float16* __restrict__ projT,
                                              const float* __restrict__ proj_b,
                                              const float* __restrict__ qin,
                                              float* __restrict__ xout) {
  __shared__ __align__(16) _Float16 smem[32768];  // 64 KB
  int h = threadIdx.x >> 5, lane = threadIdx.x & 31;
  int win = blockIdx.x;
  int b = win >> 6, wr = (win >> 3) & 7, wc = win & 7;
  int base = b * 3136 + (wr * 7) * 56 + wc * 7;   // token = base + (t/7)*56 + t%7

  _Float16* Vt = smem + h * (32 * 64);                  // [hd 32][tok 64], per head
  _Float16* Pc = smem + 8 * (32 * 64) + h * (64 * 32);  // P chunk [64][32], per head

  // stage V^T (pad tokens >= 49 with zeros); lane = head-dim element
  for (int t = 0; t < 64; ++t) {
    _Float16 val = (_Float16)0.f;
    if (t < 49) {
      int tok = base + (t / 7) * 56 + (t % 7);
      val = KV[(size_t)tok * 512 + 256 + h * 32 + lane];
    }
    Vt[lane * 64 + t] = val;
  }

  // ---- S = Q K^T (padded 64x64), 16 WMMAs, K-dim = HD = 32 ----
  f8v s[4][4] = {};
  {
    h16v amt[4];
    int m = lane & 15, hh = lane >> 4;
#pragma unroll
    for (int mt = 0; mt < 4; ++mt) {
      int t = mt * 16 + m; if (t > 48) t = 48;
      int tok = base + (t / 7) * 56 + (t % 7);
      const _Float16* r = Qh + (size_t)tok * 256 + h * 32 + hh * 16;
      amt[mt] = cat8(*(const h8v*)r, *(const h8v*)(r + 8));
    }
    int nn = lane & 15, g = lane >> 4;
#pragma unroll
    for (int nt = 0; nt < 4; ++nt) {
      int t = nt * 16 + nn; if (t > 48) t = 48;
      int tok = base + (t / 7) * 56 + (t % 7);
      const _Float16* r = KV + (size_t)tok * 512 + h * 32 + g * 16;  // B[k][n] = K[tok_n][k]
      h16v bt = cat8(*(const h8v*)r, *(const h8v*)(r + 8));
#pragma unroll
      for (int mt = 0; mt < 4; ++mt) s[mt][nt] = wmma_f16(amt[mt], bt, s[mt][nt]);
    }
  }

  // ---- + rpb, mask pad columns, softmax over rows (in registers) ----
  int n = lane & 15, hl = lane >> 4;
  const float* rpbh = rpb + h * 4096;
#pragma unroll
  for (int mt = 0; mt < 4; ++mt)
#pragma unroll
    for (int nt = 0; nt < 4; ++nt) {
      int cn = nt * 16 + n;
#pragma unroll
      for (int i = 0; i < 8; ++i) {
        int rm = mt * 16 + hl * 8 + i;
        float v = s[mt][nt][i] + rpbh[rm * 64 + cn];
        if (cn >= 49) v = -1e30f;
        s[mt][nt][i] = v;
      }
    }
#pragma unroll
  for (int mt = 0; mt < 4; ++mt)
#pragma unroll
    for (int i = 0; i < 8; ++i) {
      float mx = -1e30f;
#pragma unroll
      for (int nt = 0; nt < 4; ++nt) mx = fmaxf(mx, s[mt][nt][i]);
      for (int sh = 8; sh; sh >>= 1) mx = fmaxf(mx, __shfl_xor(mx, sh, 32));
      float sum = 0.f;
#pragma unroll
      for (int nt = 0; nt < 4; ++nt) { float e = __expf(s[mt][nt][i] - mx); s[mt][nt][i] = e; sum += e; }
      for (int sh = 8; sh; sh >>= 1) sum += __shfl_xor(sum, sh, 32);
      float inv = 1.f / sum;
#pragma unroll
      for (int nt = 0; nt < 4; ++nt) s[mt][nt][i] *= inv;
    }

  // ---- O = P @ V, two K-chunks of 32 tokens via LDS-staged P ----
  f8v o[4][2] = {};
#pragma unroll
  for (int kc = 0; kc < 2; ++kc) {
#pragma unroll
    for (int mt = 0; mt < 4; ++mt)
#pragma unroll
      for (int ntc = 0; ntc < 2; ++ntc) {
        int nt = kc * 2 + ntc;
#pragma unroll
        for (int i = 0; i < 8; ++i)
          Pc[(mt * 16 + hl * 8 + i) * 32 + ntc * 16 + n] = (_Float16)s[mt][nt][i];
      }
    // DS ops are in-order within a wave; wave-private region, no barrier needed.
#pragma unroll
    for (int mt = 0; mt < 4; ++mt) {
      h16v a = load_a(Pc + mt * 16 * 32, 32, lane);
#pragma unroll
      for (int nh2 = 0; nh2 < 2; ++nh2) {
        const _Float16* r = Vt + (size_t)(nh2 * 16 + (lane & 15)) * 64 + kc * 32 + (lane >> 4) * 16;
        h16v bt = cat8(*(const h8v*)r, *(const h8v*)(r + 8));
        o[mt][nh2] = wmma_f16(a, bt, o[mt][nh2]);
      }
    }
  }

  // ---- gather heads to LDS, fused projection, bias + residual, write x ----
  __syncthreads();
  _Float16* Os = smem;  // [64][256] f16 = 32 KB
#pragma unroll
  for (int mt = 0; mt < 4; ++mt)
#pragma unroll
    for (int nh2 = 0; nh2 < 2; ++nh2)
#pragma unroll
      for (int i = 0; i < 8; ++i)
        Os[(mt * 16 + hl * 8 + i) * 256 + h * 32 + nh2 * 16 + n] = (_Float16)o[mt][nh2][i];
  __syncthreads();

  f8v pacc[4][2] = {};
  for (int kc = 0; kc < 256; kc += 32) {
    h16v as[4];
#pragma unroll
    for (int mt = 0; mt < 4; ++mt) as[mt] = load_a(Os + (size_t)(mt * 16) * 256 + kc, 256, lane);
#pragma unroll
    for (int nt = 0; nt < 2; ++nt) {
      h16v bt = load_bT(projT + (size_t)(h * 32 + nt * 16) * 256 + kc, 256, lane);
#pragma unroll
      for (int mt = 0; mt < 4; ++mt) pacc[mt][nt] = wmma_f16(as[mt], bt, pacc[mt][nt]);
    }
  }
#pragma unroll
  for (int mt = 0; mt < 4; ++mt)
#pragma unroll
    for (int i = 0; i < 8; ++i) {
      int t = mt * 16 + hl * 8 + i;
      if (t < 49) {
        int tok = base + (t / 7) * 56 + (t % 7);
#pragma unroll
        for (int nt = 0; nt < 2; ++nt) {
          int c = h * 32 + nt * 16 + n;
          xout[(size_t)tok * 256 + c] = qin[(size_t)tok * 256 + c] + proj_b[c] + pacc[mt][nt][i];
        }
      }
    }
}

// ---------------- host launch ----------------
extern "C" void kernel_launch(void* const* d_in, const int* in_sizes, int n_in,
                              void* d_out, int out_size, void* d_ws, size_t ws_size,
                              hipStream_t stream) {
  (void)in_sizes; (void)n_in; (void)out_size; (void)ws_size;
  const float* q = (const float*)d_in[0];
  const float* k = (const float*)d_in[1];
  // d_in[2]=H, d_in[3]=W (56, hardcoded). params in dict order:
  int p = 4;
  const float* norm_q_g = (const float*)d_in[p++]; const float* norm_q_b = (const float*)d_in[p++];
  const float* norm_k_g = (const float*)d_in[p++]; const float* norm_k_b = (const float*)d_in[p++];
  const float* q_w  = (const float*)d_in[p++];     const float* q_b  = (const float*)d_in[p++];
  const float* kv_w = (const float*)d_in[p++];     const float* kv_b = (const float*)d_in[p++];
  const float* proj_w = (const float*)d_in[p++];   const float* proj_b = (const float*)d_in[p++];
  const float* pos_proj_w = (const float*)d_in[p++]; const float* pos_proj_b = (const float*)d_in[p++];
  const float* pos1_g = (const float*)d_in[p++]; const float* pos1_b = (const float*)d_in[p++];
  const float* pos1_w = (const float*)d_in[p++]; const float* pos1_bb = (const float*)d_in[p++];
  const float* pos2_g = (const float*)d_in[p++]; const float* pos2_b = (const float*)d_in[p++];
  const float* pos2_w = (const float*)d_in[p++]; const float* pos2_bb = (const float*)d_in[p++];
  const float* pos3_g = (const float*)d_in[p++]; const float* pos3_b = (const float*)d_in[p++];
  const float* pos3_w = (const float*)d_in[p++]; const float* pos3_bb = (const float*)d_in[p++];
  const float* norm2_g = (const float*)d_in[p++]; const float* norm2_b = (const float*)d_in[p++];
  const float* fc1_w = (const float*)d_in[p++]; const float* fc1_b = (const float*)d_in[p++];
  const float* fc2_w = (const float*)d_in[p++]; const float* fc2_b = (const float*)d_in[p++];

  char* ws = (char*)d_ws;
  const size_t S1 = (size_t)TTOK * CDIM * 2;  // one f16 activation plane (51.4 MB)
  float*    rpb   = (float*)(ws + 0);                        // 128 KB (padded 8x64x64)
  _Float16* qwT   = (_Float16*)(ws + (131072));
  _Float16* kvwT  = (_Float16*)(ws + (131072 + 131072));
  _Float16* projT = (_Float16*)(ws + (131072 + 131072 + 262144));
  _Float16* fc1T  = (_Float16*)(ws + (131072 + 131072 + 262144 + 131072));
  _Float16* fc2T  = (_Float16*)(ws + (131072 + 131072 + 262144 + 131072 + 524288));
  const size_t off = 2ull << 20;
  _Float16* qn  = (_Float16*)(ws + off);
  _Float16* kn  = (_Float16*)(ws + off + S1);
  _Float16* Qh  = (_Float16*)(ws + off + 2 * S1);
  _Float16* KV  = (_Float16*)(ws + off + 3 * S1);           // [tok][512]: K cols 0-255, V cols 256-511
  float*    xb  = (float*)(ws + off);                        // overlays qn+kn (free after projections)
  _Float16* lnx = (_Float16*)(ws + off + 2 * S1);            // overlays Qh (free after attention)
  _Float16* h1  = (_Float16*)(ws + off + 3 * S1);            // overlays KV + tail (205.5 MB)

  k_prep_weights<<<1024, 256, 0, stream>>>(q_w, kv_w, proj_w, fc1_w, fc2_w,
                                           qwT, kvwT, projT, fc1T, fc2T);
  k_posbias<<<1, 256, 0, stream>>>(pos_proj_w, pos_proj_b,
                                   pos1_g, pos1_b, pos1_w, pos1_bb,
                                   pos2_g, pos2_b, pos2_w, pos2_bb,
                                   pos3_g, pos3_b, pos3_w, pos3_bb, rpb);
  k_layernorm<<<TTOK / 8, 256, 0, stream>>>(q, norm_q_g, norm_q_b, qn);
  k_layernorm<<<TTOK / 8, 256, 0, stream>>>(k, norm_k_g, norm_k_b, kn);

  dim3 gq(TTOK / 128, 4);
  k_gemm<<<gq, 256, 0, stream>>>(qn, qwT, q_b, 256, 256, SCALE, 0, Qh, nullptr, nullptr);
  dim3 gkv(TTOK / 128, 8);
  k_gemm<<<gkv, 256, 0, stream>>>(kn, kvwT, kv_b, 512, 256, 1.f, 0, KV, nullptr, nullptr);

  k_attn<<<NWINS, 256, 0, stream>>>(Qh, KV, rpb, projT, proj_b, q, xb);

  k_layernorm<<<TTOK / 8, 256, 0, stream>>>(xb, norm2_g, norm2_b, lnx);
  dim3 g1(TTOK / 128, 16);
  k_gemm<<<g1, 256, 0, stream>>>(lnx, fc1T, fc1_b, 1024, 256, 1.f, 1, h1, nullptr, nullptr);
  dim3 g2(TTOK / 128, 4);
  k_gemm<<<g2, 256, 0, stream>>>(h1, fc2T, fc2_b, 256, 1024, 1.f, 2, nullptr, xb, (float*)d_out);
}
